// CGAT_62697932587340
// MI455X (gfx1250) — matrix operand order repaired and checked by reading, not compiled, not run on hardware
//
#include <hip/hip_runtime.h>
#include <hip/hip_bf16.h>

typedef __attribute__((ext_vector_type(16))) _Float16 v16h;
typedef __attribute__((ext_vector_type(8)))  _Float16 v8h;
typedef __attribute__((ext_vector_type(8)))  float    v8f;

#define DD    64
#define BB    1024
#define MM    32
#define KKN   16
#define LLP   8
#define NROWS (BB*MM)     // 32768
#define NKGT  65536
#define NENT  200000
#define NRELV 65

// ---------------- WMMA helpers (v_wmma_f32_16x16x32_f16, wave32) ----------------

static __device__ __forceinline__ v8f vzero8(){
  v8f z;
#pragma unroll
  for (int i = 0; i < 8; ++i) z[i] = 0.f;
  return z;
}

static __device__ __forceinline__ v8f wmma16(v16h a, v16h b, v8f c){
  // 8 args: (neg_a, A, neg_b, B, c_mod, C, reuse_a, reuse_b)
  return __builtin_amdgcn_wmma_f32_16x16x32_f16(false, a, false, b, (short)0, c, false, false);
}

// Async global->LDS 16B chunk (ASYNCcnt-tracked; generic-ptr low 32 bits = LDS offset).
static __device__ __forceinline__ void async_b128(void* ldst, const void* src){
  const unsigned d = (unsigned)(unsigned long long)ldst;
  asm volatile("global_load_async_to_lds_b128 %0, %1, off" :: "v"(d), "v"(src) : "memory");
}
static __device__ __forceinline__ void async_wait(){
  asm volatile("s_wait_asynccnt 0x0" ::: "memory");
}

// A fragment (16x32 f16) from row-major f16 array (LDS or global).
static __device__ __forceinline__ v16h frag_a_h(const _Float16* A, int lda, int m0, int k0, int lane){
  const int m  = m0 + (lane & 15);
  const int kb = k0 + ((lane < 16) ? 0 : 8);
  const _Float16* p = A + (size_t)m * lda + kb;
  v8h lo = *(const v8h*)(p);
  v8h hi = *(const v8h*)(p + 16);
  v16h a;
#pragma unroll
  for (int i = 0; i < 8; ++i){ a[i] = lo[i]; a[8 + i] = hi[i]; }
  return a;
}

// A fragment where every row of the tile is the same f16 vector v.
static __device__ __forceinline__ v16h frag_a_bcast_h(const _Float16* v, int k0, int lane){
  const int kb = k0 + ((lane < 16) ? 0 : 8);
  v8h lo = *(const v8h*)(v + kb);
  v8h hi = *(const v8h*)(v + kb + 16);
  v16h a;
#pragma unroll
  for (int i = 0; i < 8; ++i){ a[i] = lo[i]; a[8 + i] = hi[i]; }
  return a;
}

// B fragment (32x16 f16): B = W^T, W row-major (N x Kdim) pre-converted f16.
static __device__ __forceinline__ v16h frag_b_h(const _Float16* W, int ldk, int n0, int k0, int lane){
  const int n  = n0 + (lane & 15);
  const int kb = k0 + ((lane < 16) ? 0 : 8);
  const _Float16* p = W + (size_t)n * ldk + kb;
  v8h lo = *(const v8h*)(p);
  v8h hi = *(const v8h*)(p + 16);
  v16h b;
#pragma unroll
  for (int i = 0; i < 8; ++i){ b[i] = lo[i]; b[8 + i] = hi[i]; }
  return b;
}

// C/D store: VGPR r -> row m0 + r + (lane<16 ? 0 : 8), col n0 + (lane&15).
static __device__ __forceinline__ void frag_store(float* C, int ldc, int m0, int n0, int lane, v8f c){
  const int n  = n0 + (lane & 15);
  const int mo = m0 + ((lane < 16) ? 0 : 8);
#pragma unroll
  for (int r = 0; r < 8; ++r) C[(size_t)(mo + r) * ldc + n] = c[r];
}

static __device__ __forceinline__ void frag_store_h(_Float16* C, int ldc, int m0, int n0, int lane, v8f c){
  const int n  = n0 + (lane & 15);
  const int mo = m0 + ((lane < 16) ? 0 : 8);
#pragma unroll
  for (int r = 0; r < 8; ++r) C[(size_t)(mo + r) * ldc + n] = (_Float16)c[r];
}

static __device__ __forceinline__ float logsigf(float x){
  return (x >= 0.f) ? -log1pf(__expf(-x)) : (x - log1pf(__expf(x)));
}
static __device__ __forceinline__ float sigmoidf_(float x){
  return 1.f / (1.f + __expf(-x));
}

// ---------------- glue kernels ----------------

__global__ void k_zero(float* acc){
  if (threadIdx.x < 2) acc[threadIdx.x] = 0.f;
}

__global__ void k_cvt(const float* __restrict__ src, _Float16* __restrict__ dst, int n){
  const int i = blockIdx.x * 256 + threadIdx.x;
  if (i < n) dst[i] = (_Float16)src[i];
}

__global__ void k_user_gather(const int* __restrict__ uidx, const float* __restrict__ user_emb,
                              float* __restrict__ user_g){
  int i = blockIdx.x * 256 + threadIdx.x;
  if (i < BB * DD) user_g[i] = user_emb[(size_t)uidx[i >> 6] * DD + (i & 63)];
}

// u_att_g = relu(user_g @ W_tatt^T + b_tatt)   (1024x64 @ 64x64)  -- one wave, 16 rows/block
__global__ void k_uatt(const float* __restrict__ user_g, const _Float16* __restrict__ hW_tatt,
                       const float* __restrict__ b_tatt, float* __restrict__ uatt_g){
  __shared__ _Float16 sA[16 * DD];
  const int lane = threadIdx.x;            // blockDim = 32
  const int row16 = blockIdx.x * 16;
  __builtin_prefetch(hW_tatt, 0, 0);
  for (int i = lane; i < 16 * DD; i += 32) sA[i] = (_Float16)user_g[(size_t)row16 * DD + i];
  __syncthreads();
  v8f acc[4];
#pragma unroll
  for (int t = 0; t < 4; ++t) acc[t] = vzero8();
  for (int k0 = 0; k0 < 64; k0 += 32){
    v16h a = frag_a_h(sA, DD, 0, k0, lane);
#pragma unroll
    for (int t = 0; t < 4; ++t){
      v16h b = frag_b_h(hW_tatt, 64, t * 16, k0, lane);
      acc[t] = wmma16(a, b, acc[t]);
    }
  }
#pragma unroll
  for (int t = 0; t < 4; ++t){
    const int n  = t * 16 + (lane & 15);
    const int mo = (lane < 16) ? 0 : 8;
#pragma unroll
    for (int r = 0; r < 8; ++r)
      uatt_g[(size_t)(row16 + mo + r) * DD + n] = fmaxf(acc[t][r] + b_tatt[n], 0.f);
  }
}

// ---------------- local-connection aggregator (agg_lc) ----------------
// block = 4 output rows (64 A-rows = 4 neighbor tiles), 128 threads (4 waves).
#define LC_SA    0                         // f16 64x128
#define LC_SNERE (LC_SA + 64*128*2)        // f16 64x64
#define LC_SA3   (LC_SNERE + 64*64*2)      // f16 16x128
#define LC_SSH   (LC_SA3 + 16*128*2)       // f16 4x64 (self, f16 copy)
#define LC_STR   (LC_SSH + 4*64*2)         // f32 64x64
#define LC_SSELF (LC_STR + 64*64*4)        // f32 4x64
#define LC_SU    (LC_SSELF + 4*64*4)       // f32 4x64
#define LC_SAGG  (LC_SU + 4*64*4)          // f32 16x64
#define LC_SLOG  (LC_SAGG + 16*64*4)       // f32 64
#define LC_SATT  (LC_SLOG + 64*4)          // f32 64
#define LC_BYTES (LC_SATT + 64*4)

__global__ void k_agg_lc(int nRowsTot, int udiv,
    const int* __restrict__ self_idx, const int* __restrict__ nbr_idx, const int* __restrict__ rel_idx,
    const float* __restrict__ entemb,
    const _Float16* __restrict__ ent16, const _Float16* __restrict__ rel16,
    const float* __restrict__ uatt_g,
    const _Float16* __restrict__ hW_re, const _Float16* __restrict__ hW_gatt, const float* __restrict__ b_gatt,
    const _Float16* __restrict__ hW_agg, const float* __restrict__ b_agg,
    float* __restrict__ self_out, float* __restrict__ lc_out)
{
  extern __shared__ char smraw[];
  _Float16* sA    = (_Float16*)(smraw + LC_SA);
  _Float16* sNere = (_Float16*)(smraw + LC_SNERE);
  _Float16* sA3   = (_Float16*)(smraw + LC_SA3);
  _Float16* sSelfH= (_Float16*)(smraw + LC_SSH);
  float*    sTr   = (float*)(smraw + LC_STR);
  float*    sSelf = (float*)(smraw + LC_SSELF);
  float*    sU    = (float*)(smraw + LC_SU);
  float*    sAgg  = (float*)(smraw + LC_SAGG);
  float*    sLog  = (float*)(smraw + LC_SLOG);
  float*    sAtt  = (float*)(smraw + LC_SATT);

  const int tid  = threadIdx.x;
  const int lane = tid & 31;
  const int wv   = tid >> 5;
  const int row4 = blockIdx.x * 4;

  __builtin_prefetch(hW_re, 0, 0);
  __builtin_prefetch(hW_gatt, 0, 0);

  // async gather: neighbor + relation rows stream straight into LDS as f16 (16B chunks)
  for (int i = tid; i < 64 * 16; i += 128){
    const int ar = i >> 4, ch = i & 15;          // A-row, 16B chunk within 256B row
    const int n  = row4 + (ar >> 4);
    const int k  = ar & 15;
    const int nc = (n < nRowsTot) ? n : (nRowsTot - 1);
    if (ch < 8)
      async_b128(sA + (size_t)ar * 128 + ch * 8,
                 ent16 + (size_t)nbr_idx[(size_t)nc * KKN + k] * DD + ch * 8);
    else
      async_b128(sA + (size_t)ar * 128 + ch * 8,
                 rel16 + (size_t)rel_idx[(size_t)nc * KKN + k] * DD + (ch - 8) * 8);
  }
  for (int i = tid; i < 4 * 64; i += 128){
    const int r = i >> 6, c = i & 63;
    const int n = row4 + r;
    const int nc = (n < nRowsTot) ? n : (nRowsTot - 1);
    const float sv = (n < nRowsTot) ? entemb[(size_t)self_idx[nc] * DD + c] : 0.f;
    sSelf[i]  = sv;
    sSelfH[i] = (_Float16)sv;
    sU[i]     = (n < nRowsTot) ? uatt_g[(size_t)(nc / udiv) * DD + c] : 0.f;
  }
  async_wait();
  __syncthreads();

  // GEMM1: nere(64x64) = [nbr|rel](64x128) @ W_re^T(128x64)
  {
    const int m0 = wv * 16;
    v8f acc[4];
#pragma unroll
    for (int t = 0; t < 4; ++t) acc[t] = vzero8();
    for (int k0 = 0; k0 < 128; k0 += 32){
      v16h a = frag_a_h(sA, 128, m0, k0, lane);
#pragma unroll
      for (int t = 0; t < 4; ++t){
        v16h b = frag_b_h(hW_re, 128, t * 16, k0, lane);
        acc[t] = wmma16(a, b, acc[t]);
      }
    }
#pragma unroll
    for (int t = 0; t < 4; ++t) frag_store_h(sNere, 64, m0, t * 16, lane, acc[t]);
  }
  __syncthreads();

  // GEMM2: trans = tanh([self1|nere] @ W_gatt^T + b_gatt); tile rows share one self vector
  {
    const int m0 = wv * 16;
    v8f acc[4];
#pragma unroll
    for (int t = 0; t < 4; ++t) acc[t] = vzero8();
    for (int k0 = 0; k0 < 128; k0 += 32){
      v16h a = (k0 < 64) ? frag_a_bcast_h(sSelfH + wv * 64, k0, lane)
                         : frag_a_h(sNere, 64, m0, k0 - 64, lane);
#pragma unroll
      for (int t = 0; t < 4; ++t){
        v16h b = frag_b_h(hW_gatt, 128, t * 16, k0, lane);
        acc[t] = wmma16(a, b, acc[t]);
      }
    }
#pragma unroll
    for (int t = 0; t < 4; ++t){
      const int n  = t * 16 + (lane & 15);
      const int mo = m0 + ((lane < 16) ? 0 : 8);
#pragma unroll
      for (int r = 0; r < 8; ++r)
        sTr[(size_t)(mo + r) * 64 + n] = tanhf(acc[t][r] + b_gatt[n]);
    }
  }
  __syncthreads();

  // attention logits + softmax over K=16 neighbors
  if (tid < 64){
    const float* u = sU + (tid >> 4) * 64;
    const float* tr = sTr + tid * 64;
    float s = 0.f;
    for (int d = 0; d < 64; ++d) s += u[d] * tr[d];
    sLog[tid] = s;
  }
  __syncthreads();
  if (tid < 4){
    float mx = -1e30f;
    for (int k = 0; k < 16; ++k) mx = fmaxf(mx, sLog[tid * 16 + k]);
    float ss = 0.f;
    for (int k = 0; k < 16; ++k){ float e = __expf(sLog[tid * 16 + k] - mx); sAtt[tid * 16 + k] = e; ss += e; }
    const float inv = 1.f / ss;
    for (int k = 0; k < 16; ++k) sAtt[tid * 16 + k] *= inv;
  }
  __syncthreads();

  // A3 = [self | new_v] for rows 0..3, zero-padded to 16 rows (built in f16)
  for (int i = tid; i < 16 * 128; i += 128){
    const int r = i >> 7, c = i & 127;
    float v = 0.f;
    if (r < 4){
      if (c < 64) v = sSelf[r * 64 + c];
      else {
        float s = 0.f;
        for (int k = 0; k < 16; ++k) s += sAtt[r * 16 + k] * (float)sA[(size_t)(r * 16 + k) * 128 + (c - 64)];
        v = s;
      }
    }
    sA3[i] = (_Float16)v;
  }
  __syncthreads();

  // GEMM3: agg = tanh(A3 @ W_agg^T + b_agg)  (wave 0 only; whole-wave uniform)
  if (wv == 0){
    v8f acc[4];
#pragma unroll
    for (int t = 0; t < 4; ++t) acc[t] = vzero8();
    for (int k0 = 0; k0 < 128; k0 += 32){
      v16h a = frag_a_h(sA3, 128, 0, k0, lane);
#pragma unroll
      for (int t = 0; t < 4; ++t){
        v16h b = frag_b_h(hW_agg, 128, t * 16, k0, lane);
        acc[t] = wmma16(a, b, acc[t]);
      }
    }
#pragma unroll
    for (int t = 0; t < 4; ++t){
      const int n  = t * 16 + (lane & 15);
      const int mo = (lane < 16) ? 0 : 8;
#pragma unroll
      for (int r = 0; r < 8; ++r)
        sAgg[(size_t)(mo + r) * 64 + n] = tanhf(acc[t][r] + b_agg[n]);
    }
  }
  __syncthreads();
  if (tid < 4){
    float s = 0.f;
    for (int d = 0; d < 64; ++d){ float v = sAgg[tid * 64 + d]; s += v * v; }
    sLog[tid] = 1.f / fmaxf(sqrtf(s), 1e-12f);
  }
  __syncthreads();
  for (int i = tid; i < 4 * 64; i += 128){
    const int r = i >> 6, c = i & 63;
    const int n = row4 + r;
    if (n < nRowsTot){
      self_out[(size_t)n * DD + c] = sSelf[i];
      lc_out  [(size_t)n * DD + c] = sAgg[r * 64 + c] * sLog[r];
    }
  }
}

// ---------------- GRU path aggregator (agg_gb) ----------------
#define GR_SX    0                         // f16 16x64
#define GR_SHH   (GR_SX + 16*64*2)         // f16 16x64 (h, f16 copy)
#define GR_SSH   (GR_SHH + 16*64*2)        // f16 16x64 (self, f16)
#define GR_SH    (GR_SSH + 16*64*2)        // f32 16x64
#define GR_SGI   (GR_SH + 16*64*4)         // f32 16x192
#define GR_SGH   (GR_SGI + 16*192*4)       // f32 16x192
#define GR_SAGG  (GR_SGH + 16*192*4)       // f32 16x64
#define GR_SNRM  (GR_SAGG + 16*64*4)       // f32 16
#define GR_BYTES (GR_SNRM + 16*4)

__global__ void k_gru(int nRowsTot,
    const int* __restrict__ self_idx, const int* __restrict__ path_idx,
    const _Float16* __restrict__ ent16,
    const _Float16* __restrict__ hW_ih, const _Float16* __restrict__ hW_hh,
    const float* __restrict__ b_ih, const float* __restrict__ b_hh,
    const _Float16* __restrict__ hW_agg, const float* __restrict__ b_agg,
    float* __restrict__ gb_out)
{
  extern __shared__ char smraw[];
  _Float16* sX    = (_Float16*)(smraw + GR_SX);
  _Float16* sHh   = (_Float16*)(smraw + GR_SHH);
  _Float16* sSelfH= (_Float16*)(smraw + GR_SSH);
  float*    sH    = (float*)(smraw + GR_SH);
  float*    sGi   = (float*)(smraw + GR_SGI);
  float*    sGh   = (float*)(smraw + GR_SGH);
  float*    sAgg  = (float*)(smraw + GR_SAGG);
  float*    sNrm  = (float*)(smraw + GR_SNRM);

  const int tid  = threadIdx.x;
  const int lane = tid & 31;
  const int wv   = tid >> 5;
  const int row16 = blockIdx.x * 16;

  __builtin_prefetch(hW_ih, 0, 0);
  __builtin_prefetch(hW_hh, 0, 0);

  // self rows: async f16 gather (16 rows x 8 chunks = 128 chunks, one per thread)
  {
    const int r = tid >> 3, ch = tid & 7;
    const int n = row16 + r;
    const int nc = (n < nRowsTot) ? n : (nRowsTot - 1);
    async_b128(sSelfH + (size_t)r * 64 + ch * 8,
               ent16 + (size_t)self_idx[nc] * DD + ch * 8);
  }
  for (int i = tid; i < 16 * 64; i += 128){
    sH[i]  = 0.f;
    sHh[i] = (_Float16)0.f;
  }
  async_wait();
  __syncthreads();

  // hoist weight fragments: wave wv owns column tiles wv, wv+4, wv+8
  v16h bIH[3][2], bHH[3][2];
#pragma unroll
  for (int li = 0; li < 3; ++li){
    const int tt = wv + li * 4;
#pragma unroll
    for (int ks = 0; ks < 2; ++ks){
      bIH[li][ks] = frag_b_h(hW_ih, 64, tt * 16, ks * 32, lane);
      bHH[li][ks] = frag_b_h(hW_hh, 64, tt * 16, ks * 32, lane);
    }
  }

  for (int t = 0; t < LLP; ++t){
    {
      const int r = tid >> 3, ch = tid & 7;
      const int n = row16 + r;
      const int nc = (n < nRowsTot) ? n : (nRowsTot - 1);
      async_b128(sX + (size_t)r * 64 + ch * 8,
                 ent16 + (size_t)path_idx[(size_t)nc * LLP + t] * DD + ch * 8);
    }
    async_wait();
    __syncthreads();
#pragma unroll
    for (int li = 0; li < 3; ++li){
      const int tt = wv + li * 4;
      v8f ai = vzero8(), ah = vzero8();
#pragma unroll
      for (int ks = 0; ks < 2; ++ks){
        v16h ax = frag_a_h(sX, 64, 0, ks * 32, lane);
        ai = wmma16(ax, bIH[li][ks], ai);
        v16h axh = frag_a_h(sHh, 64, 0, ks * 32, lane);
        ah = wmma16(axh, bHH[li][ks], ah);
      }
      const int n  = tt * 16 + (lane & 15);
      const int mo = (lane < 16) ? 0 : 8;
#pragma unroll
      for (int r = 0; r < 8; ++r){
        sGi[(size_t)(mo + r) * 192 + n] = ai[r] + b_ih[n];
        sGh[(size_t)(mo + r) * 192 + n] = ah[r] + b_hh[n];
      }
    }
    __syncthreads();
    for (int i = tid; i < 16 * 64; i += 128){
      const int r = i >> 6, c = i & 63;
      const float ir = sGi[r * 192 + c],       hr = sGh[r * 192 + c];
      const float iz = sGi[r * 192 + 64 + c],  hz = sGh[r * 192 + 64 + c];
      const float in_ = sGi[r * 192 + 128 + c], hn = sGh[r * 192 + 128 + c];
      const float rg = sigmoidf_(ir + hr);
      const float zg = sigmoidf_(iz + hz);
      const float nn = tanhf(in_ + rg * hn);
      const float hv = (1.f - zg) * nn + zg * sH[i];
      sH[i]  = hv;
      sHh[i] = (_Float16)hv;
    }
    __syncthreads();
  }

  // agg = tanh([self | hT] @ W_agg^T + b_agg); wave wv owns column tile wv
  {
    v8f acc = vzero8();
    for (int k0 = 0; k0 < 128; k0 += 32){
      v16h a = (k0 < 64) ? frag_a_h(sSelfH, 64, 0, k0, lane)
                         : frag_a_h(sHh, 64, 0, k0 - 64, lane);
      v16h b = frag_b_h(hW_agg, 128, wv * 16, k0, lane);
      acc = wmma16(a, b, acc);
    }
    const int n  = wv * 16 + (lane & 15);
    const int mo = (lane < 16) ? 0 : 8;
#pragma unroll
    for (int r = 0; r < 8; ++r)
      sAgg[(size_t)(mo + r) * 64 + n] = tanhf(acc[r] + b_agg[n]);
  }
  __syncthreads();
  if (tid < 16){
    float s = 0.f;
    for (int d = 0; d < 64; ++d){ float v = sAgg[tid * 64 + d]; s += v * v; }
    sNrm[tid] = 1.f / fmaxf(sqrtf(s), 1e-12f);
  }
  __syncthreads();
  for (int i = tid; i < 16 * 64; i += 128){
    const int r = i >> 6, c = i & 63;
    const int n = row16 + r;
    if (n < nRowsTot) gb_out[(size_t)n * DD + c] = sAgg[i] * sNrm[r];
  }
}

// item = [self, g*lc + (1-g)*gb]
__global__ void k_combine(int nRowsTot, const float* __restrict__ selfv, const float* __restrict__ lc,
                          const float* __restrict__ gb, const float* __restrict__ gate,
                          float* __restrict__ out){
  const int i = blockIdx.x * 256 + threadIdx.x;
  if (i >= nRowsTot * 128) return;
  const int n = i >> 7, c = i & 127;
  float v;
  if (c < 64) v = selfv[(size_t)n * DD + c];
  else {
    const int d = c - 64;
    const float g = sigmoidf_(gate[d]);
    v = g * lc[(size_t)n * DD + d] + (1.f - g) * gb[(size_t)n * DD + d];
  }
  out[i] = v;
}

// rs_att + build A = [user_g | u_lc] (written as f16 for the user MLP GEMM).
__global__ void k_rs_att(const float* __restrict__ item_hist, const float* __restrict__ item,
                         const float* __restrict__ user_g, const float* __restrict__ W_iatt,
                         const float* __restrict__ b_iatt, _Float16* __restrict__ amlp){
  __shared__ float sItem[128], sRed[128], sLog[32], sAtt[32];
  const int b = blockIdx.x, tid = threadIdx.x;
  sItem[tid] = item[(size_t)b * 128 + tid];
  __syncthreads();
  sRed[tid] = W_iatt[128 + tid] * sItem[tid];
  __syncthreads();
  for (int s = 64; s > 0; s >>= 1){ if (tid < s) sRed[tid] += sRed[tid + s]; __syncthreads(); }
  const float cb = sRed[0] + b_iatt[0];
  if (tid < 32){
    const float* h = item_hist + ((size_t)b * MM + tid) * 128;
    float s = cb;
    for (int d = 0; d < 128; ++d) s += W_iatt[d] * h[d];
    sLog[tid] = tanhf(s);
  }
  __syncthreads();
  if (tid == 0){
    float mx = -1e30f;
    for (int m = 0; m < 32; ++m) mx = fmaxf(mx, sLog[m]);
    float ss = 0.f;
    for (int m = 0; m < 32; ++m){ float e = __expf(sLog[m] - mx); sAtt[m] = e; ss += e; }
    const float inv = 1.f / ss;
    for (int m = 0; m < 32; ++m) sAtt[m] *= inv;
  }
  __syncthreads();
  float ul = 0.f;
  for (int m = 0; m < 32; ++m) ul += sAtt[m] * item_hist[((size_t)b * MM + m) * 128 + tid];
  if (tid < 64) amlp[(size_t)b * 192 + tid] = (_Float16)user_g[(size_t)b * DD + tid];
  amlp[(size_t)b * 192 + 64 + tid] = (_Float16)ul;
}

// u = relu([user_g|u_lc] @ W_u_mlp^T + b); normalize; score = <[user_g, u_n], item>.
__global__ void k_umlp_score(const _Float16* __restrict__ amlp, const _Float16* __restrict__ hW_u,
                             const float* __restrict__ b_u, const float* __restrict__ user_g,
                             const float* __restrict__ item, float* __restrict__ score_out){
  __shared__ float sU[16 * 64];
  __shared__ float sNrm[16];
  const int lane = threadIdx.x;          // blockDim = 32
  const int row16 = blockIdx.x * 16;
  __builtin_prefetch(hW_u, 0, 0);
  v8f acc[4];
#pragma unroll
  for (int t = 0; t < 4; ++t) acc[t] = vzero8();
  const _Float16* Abase = amlp + (size_t)row16 * 192;
  for (int k0 = 0; k0 < 192; k0 += 32){
    v16h a = frag_a_h(Abase, 192, 0, k0, lane);
#pragma unroll
    for (int t = 0; t < 4; ++t){
      v16h b = frag_b_h(hW_u, 192, t * 16, k0, lane);
      acc[t] = wmma16(a, b, acc[t]);
    }
  }
#pragma unroll
  for (int t = 0; t < 4; ++t){
    const int n  = t * 16 + (lane & 15);
    const int mo = (lane < 16) ? 0 : 8;
#pragma unroll
    for (int r = 0; r < 8; ++r)
      sU[(size_t)(mo + r) * 64 + n] = fmaxf(acc[t][r] + b_u[n], 0.f);
  }
  __syncthreads();
  if (lane < 16){
    float s = 0.f;
    for (int d = 0; d < 64; ++d){ float v = sU[lane * 64 + d]; s += v * v; }
    sNrm[lane] = 1.f / fmaxf(sqrtf(s), 1e-12f);
  }
  __syncthreads();
  if (lane < 16){
    const int n = row16 + lane;
    float s = 0.f;
    for (int d = 0; d < 64; ++d){
      s += user_g[(size_t)n * DD + d] * item[(size_t)n * 128 + d];
      s += sU[lane * 64 + d] * sNrm[lane] * item[(size_t)n * 128 + 64 + d];
    }
    score_out[n] = s;
  }
}

// ---------------- KG loss ----------------
// block = 64 threads (wave0 -> pos GEMM, wave1 -> neg GEMM), 16 triples/block.
// All gathers stream async into LDS (f16 tails/rels, f32 heads).
#define KG_SA    0                         // f16 16x128
#define KG_SAN   (KG_SA + 16*128*2)        // f16 16x128
#define KG_SHEAD (KG_SAN + 16*128*2)       // f32 16x64
#define KG_SNE   (KG_SHEAD + 16*64*4)      // f32 16x64
#define KG_SNN   (KG_SNE + 16*64*4)        // f32 16x64
#define KG_SRED  (KG_SNN + 16*64*4)        // f32 16
#define KG_BYTES (KG_SRED + 16*4)

__global__ void k_kg(const int* __restrict__ kg_head, const int* __restrict__ kg_rel,
                     const int* __restrict__ kg_tail, const int* __restrict__ kg_tail_neg,
                     const float* __restrict__ entemb,
                     const _Float16* __restrict__ ent16, const _Float16* __restrict__ rel16,
                     const _Float16* __restrict__ hW_re, float* __restrict__ acc){
  extern __shared__ char smraw[];
  _Float16* sA    = (_Float16*)(smraw + KG_SA);
  _Float16* sAn   = (_Float16*)(smraw + KG_SAN);
  float*    sHead = (float*)(smraw + KG_SHEAD);
  float*    sNe   = (float*)(smraw + KG_SNE);
  float*    sNn   = (float*)(smraw + KG_SNN);
  float*    sRed  = (float*)(smraw + KG_SRED);
  const int tid = threadIdx.x, lane = tid & 31, wv = tid >> 5;
  const int base = blockIdx.x * 16;
  __builtin_prefetch(hW_re, 0, 0);

  // heads (f32): 16 rows x 16 chunks of 16B
  for (int i = tid; i < 16 * 16; i += 64){
    const int r = i >> 4, ch = i & 15;
    async_b128(sHead + (size_t)r * 64 + ch * 4,
               entemb + (size_t)kg_head[base + r] * DD + ch * 4);
  }
  // tails + relations (f16): 16 rows x 16 chunks each for pos and neg
  for (int i = tid; i < 16 * 16; i += 64){
    const int r = i >> 4, ch = i & 15;
    if (ch < 8){
      async_b128(sA  + (size_t)r * 128 + ch * 8, ent16 + (size_t)kg_tail[base + r] * DD + ch * 8);
      async_b128(sAn + (size_t)r * 128 + ch * 8, ent16 + (size_t)kg_tail_neg[base + r] * DD + ch * 8);
    } else {
      const _Float16* rsrc = rel16 + (size_t)kg_rel[base + r] * DD + (ch - 8) * 8;
      async_b128(sA  + (size_t)r * 128 + ch * 8, rsrc);
      async_b128(sAn + (size_t)r * 128 + ch * 8, rsrc);
    }
  }
  async_wait();
  __syncthreads();
  {
    const _Float16* Asrc = (wv == 0) ? sA : sAn;
    float* Dst           = (wv == 0) ? sNe : sNn;
    v8f a4[4];
#pragma unroll
    for (int t = 0; t < 4; ++t) a4[t] = vzero8();
    for (int k0 = 0; k0 < 128; k0 += 32){
      v16h a = frag_a_h(Asrc, 128, 0, k0, lane);
#pragma unroll
      for (int t = 0; t < 4; ++t){
        v16h b = frag_b_h(hW_re, 128, t * 16, k0, lane);
        a4[t] = wmma16(a, b, a4[t]);
      }
    }
#pragma unroll
    for (int t = 0; t < 4; ++t) frag_store(Dst, 64, 0, t * 16, lane, a4[t]);
  }
  __syncthreads();
  if (tid < 16){
    float s = 0.f, sn = 0.f;
    for (int d = 0; d < 64; ++d){
      const float h = sHead[tid * 64 + d];
      const float a1 = h - sNe[tid * 64 + d];
      const float a2 = h - sNn[tid * 64 + d];
      s += a1 * a1; sn += a2 * a2;
    }
    sRed[tid] = logsigf(sn - s);
  }
  __syncthreads();
  if (tid == 0){
    float s = 0.f;
    for (int k = 0; k < 16; ++k) s += sRed[k];
    atomicAdd(&acc[1], s);
  }
}

__global__ void k_rsloss(const float* __restrict__ scores, float* __restrict__ acc){
  __shared__ float sRed[256];
  const int tid = threadIdx.x;
  float s = 0.f;
  for (int i = tid; i < BB; i += 256) s += logsigf(scores[i] - scores[BB + i]);
  sRed[tid] = s;
  __syncthreads();
  for (int st = 128; st > 0; st >>= 1){ if (tid < st) sRed[tid] += sRed[tid + st]; __syncthreads(); }
  if (tid == 0) atomicAdd(&acc[0], sRed[0]);
}

__global__ void k_final(const float* __restrict__ acc, float* __restrict__ out){
  if (threadIdx.x == 0 && blockIdx.x == 0)
    out[2 * BB] = -acc[0] / (float)BB - acc[1] / (float)NKGT;
}

// ---------------- host-side orchestration ----------------

extern "C" void kernel_launch(void* const* d_in, const int* in_sizes, int n_in,
                              void* d_out, int out_size, void* d_ws, size_t ws_size,
                              hipStream_t stream){
  (void)in_sizes; (void)n_in; (void)out_size; (void)ws_size;

  const int*   user_indices = (const int*)d_in[0];
  const int*   ent_his_self = (const int*)d_in[1];
  const int*   ent_his_nbr  = (const int*)d_in[2];
  const int*   rel_his      = (const int*)d_in[3];
  const int*   ent_pos_self = (const int*)d_in[4];
  const int*   ent_pos_nbr  = (const int*)d_in[5];
  const int*   rel_pos      = (const int*)d_in[6];
  const int*   ent_neg_self = (const int*)d_in[7];
  const int*   ent_neg_nbr  = (const int*)d_in[8];
  const int*   rel_neg      = (const int*)d_in[9];
  const int*   gb_his_self  = (const int*)d_in[10];
  const int*   gb_his_path  = (const int*)d_in[11];
  const int*   gb_pos_self  = (const int*)d_in[12];
  const int*   gb_pos_path  = (const int*)d_in[13];
  const int*   gb_neg_self  = (const int*)d_in[14];
  const int*   gb_neg_path  = (const int*)d_in[15];
  const int*   kg_head      = (const int*)d_in[16];
  const int*   kg_rel       = (const int*)d_in[17];
  const int*   kg_tail      = (const int*)d_in[18];
  const int*   kg_tail_neg  = (const int*)d_in[19];
  const float* entity_emb   = (const float*)d_in[20];
  const float* relation_emb = (const float*)d_in[21];
  const float* user_emb     = (const float*)d_in[22];
  const float* gate         = (const float*)d_in[23];
  const float* W_u_mlp      = (const float*)d_in[24];
  const float* b_u_mlp      = (const float*)d_in[25];
  const float* W_re         = (const float*)d_in[26];
  const float* W_agg        = (const float*)d_in[27];
  const float* b_agg        = (const float*)d_in[28];
  const float* W_gatt       = (const float*)d_in[29];
  const float* b_gatt       = (const float*)d_in[30];
  const float* W_iatt       = (const float*)d_in[31];
  const float* b_iatt       = (const float*)d_in[32];
  const float* W_tatt       = (const float*)d_in[33];
  const float* b_tatt       = (const float*)d_in[34];
  const float* W_ih         = (const float*)d_in[35];
  const float* W_hh         = (const float*)d_in[36];
  const float* b_ih         = (const float*)d_in[37];
  const float* b_hh         = (const float*)d_in[38];

  float* out = (float*)d_out;

  float* ws = (float*)d_ws;
  size_t off = 0;
  auto take = [&](size_t n){ float* p = ws + off; off += n; return p; };
  float* user_g    = take((size_t)BB * DD);
  float* uatt_g    = take((size_t)BB * DD);
  float* self_his  = take((size_t)NROWS * DD);
  float* lc_his    = take((size_t)NROWS * DD);
  float* gb_his    = take((size_t)NROWS * DD);
  float* item_hist = take((size_t)NROWS * 2 * DD);
  float* self_pos  = take((size_t)BB * DD);
  float* lc_pos    = take((size_t)BB * DD);
  float* gb_pos    = take((size_t)BB * DD);
  float* item_pos  = take((size_t)BB * 2 * DD);
  float* self_neg  = take((size_t)BB * DD);
  float* lc_neg    = take((size_t)BB * DD);
  float* gb_neg    = take((size_t)BB * DD);
  float* item_neg  = take((size_t)BB * 2 * DD);
  float* accv      = take(2);
  // f16 region (32B aligned since 'off' is rounded up to 8 floats)
  off += (8 - (off & 7)) & 7;
  _Float16* hbase = (_Float16*)(ws + off);
  size_t hoff = 0;
  auto takeh = [&](size_t n){ _Float16* p = hbase + hoff; hoff += n; return p; };
  _Float16* hW_re   = takeh((size_t)64 * 128);
  _Float16* hW_gatt = takeh((size_t)64 * 128);
  _Float16* hW_agg  = takeh((size_t)64 * 128);
  _Float16* hW_tatt = takeh((size_t)64 * 64);
  _Float16* hW_u    = takeh((size_t)64 * 192);
  _Float16* hW_ih   = takeh((size_t)192 * 64);
  _Float16* hW_hh   = takeh((size_t)192 * 64);
  _Float16* amlp_pos= takeh((size_t)BB * 192);
  _Float16* amlp_neg= takeh((size_t)BB * 192);
  _Float16* ent16   = takeh((size_t)NENT * DD);
  _Float16* rel16   = takeh((size_t)NRELV * DD);

  k_zero<<<1, 32, 0, stream>>>(accv);
  // one-time f32 -> f16 conversions (weights + embedding tables; L2-resident afterwards)
  k_cvt<<<(64*128 + 255)/256, 256, 0, stream>>>(W_re,   hW_re,   64*128);
  k_cvt<<<(64*128 + 255)/256, 256, 0, stream>>>(W_gatt, hW_gatt, 64*128);
  k_cvt<<<(64*128 + 255)/256, 256, 0, stream>>>(W_agg,  hW_agg,  64*128);
  k_cvt<<<(64*64  + 255)/256, 256, 0, stream>>>(W_tatt, hW_tatt, 64*64);
  k_cvt<<<(64*192 + 255)/256, 256, 0, stream>>>(W_u_mlp,hW_u,    64*192);
  k_cvt<<<(192*64 + 255)/256, 256, 0, stream>>>(W_ih,   hW_ih,   192*64);
  k_cvt<<<(192*64 + 255)/256, 256, 0, stream>>>(W_hh,   hW_hh,   192*64);
  k_cvt<<<(NENT*DD + 255)/256, 256, 0, stream>>>(entity_emb,   ent16, NENT*DD);
  k_cvt<<<(NRELV*DD + 255)/256, 256, 0, stream>>>(relation_emb, rel16, NRELV*DD);

  k_user_gather<<<(BB * DD + 255) / 256, 256, 0, stream>>>(user_indices, user_emb, user_g);
  k_uatt<<<BB / 16, 32, 0, stream>>>(user_g, hW_tatt, b_tatt, uatt_g);

  // history branch (N = B*M rows)
  k_agg_lc<<<NROWS / 4, 128, LC_BYTES, stream>>>(NROWS, MM, ent_his_self, ent_his_nbr, rel_his,
      entity_emb, ent16, rel16, uatt_g, hW_re, hW_gatt, b_gatt, hW_agg, b_agg, self_his, lc_his);
  k_gru<<<NROWS / 16, 128, GR_BYTES, stream>>>(NROWS, gb_his_self, gb_his_path, ent16,
      hW_ih, hW_hh, b_ih, b_hh, hW_agg, b_agg, gb_his);
  k_combine<<<(NROWS * 128 + 255) / 256, 256, 0, stream>>>(NROWS, self_his, lc_his, gb_his, gate, item_hist);

  // positive branch
  k_agg_lc<<<BB / 4, 128, LC_BYTES, stream>>>(BB, 1, ent_pos_self, ent_pos_nbr, rel_pos,
      entity_emb, ent16, rel16, uatt_g, hW_re, hW_gatt, b_gatt, hW_agg, b_agg, self_pos, lc_pos);
  k_gru<<<BB / 16, 128, GR_BYTES, stream>>>(BB, gb_pos_self, gb_pos_path, ent16,
      hW_ih, hW_hh, b_ih, b_hh, hW_agg, b_agg, gb_pos);
  k_combine<<<(BB * 128 + 255) / 256, 256, 0, stream>>>(BB, self_pos, lc_pos, gb_pos, gate, item_pos);
  k_rs_att<<<BB, 128, 0, stream>>>(item_hist, item_pos, user_g, W_iatt, b_iatt, amlp_pos);
  k_umlp_score<<<BB / 16, 32, 0, stream>>>(amlp_pos, hW_u, b_u_mlp, user_g, item_pos, out);

  // negative branch
  k_agg_lc<<<BB / 4, 128, LC_BYTES, stream>>>(BB, 1, ent_neg_self, ent_neg_nbr, rel_neg,
      entity_emb, ent16, rel16, uatt_g, hW_re, hW_gatt, b_gatt, hW_agg, b_agg, self_neg, lc_neg);
  k_gru<<<BB / 16, 128, GR_BYTES, stream>>>(BB, gb_neg_self, gb_neg_path, ent16,
      hW_ih, hW_hh, b_ih, b_hh, hW_agg, b_agg, gb_neg);
  k_combine<<<(BB * 128 + 255) / 256, 256, 0, stream>>>(BB, self_neg, lc_neg, gb_neg, gate, item_neg);
  k_rs_att<<<BB, 128, 0, stream>>>(item_hist, item_neg, user_g, W_iatt, b_iatt, amlp_neg);
  k_umlp_score<<<BB / 16, 32, 0, stream>>>(amlp_neg, hW_u, b_u_mlp, user_g, item_neg, out + BB);

  // KG loss + final combine
  k_kg<<<NKGT / 16, 64, KG_BYTES, stream>>>(kg_head, kg_rel, kg_tail, kg_tail_neg,
      entity_emb, ent16, rel16, hW_re, accv);
  k_rsloss<<<1, 256, 0, stream>>>(out, accv);
  k_final<<<1, 32, 0, stream>>>(accv, out);
}
